// ResidualRefinementSeparator_45990509806107
// MI455X (gfx1250) — matrix operand order repaired and checked by reading, not compiled, not run on hardware
//
#include <hip/hip_runtime.h>
#include <hip/hip_bf16.h>

typedef __attribute__((ext_vector_type(16))) _Float16 v16h;
typedef __attribute__((ext_vector_type(8)))  float    v8f;

#define B_TOT   262144
#define L_DIM   12
#define P_DIM   4
#define H_DIM   64
#define S_DIM   3
#define EPSV    1e-8f

#define WAVES        8
#define TILE         16
#define BLK_SAMPLES  (WAVES * TILE)      // 128 samples per block
#define COMBO_HALVES 7168                // halves per (s,p,c) weight-fragment pack
#define L1_OFF       0                   // 4 frags  (ntile 0..3), K=32 (24 padded)
#define L2_OFF       2048                // 8 frags  (kstep*4 + ntile)
#define L3_OFF       6144                // 2 frags  (kstep), N=16 (12 padded)

union FragU { uint4 q[2]; v16h v; };

// Compiler-only fence: LDS staging buffers are wave-private and the LDS unit
// processes same-wave DS ops in order, so no workgroup barrier is needed —
// only prevent the compiler from moving DS ops across the fence.
__device__ __forceinline__ void lds_fence() { __builtin_amdgcn_wave_barrier(); }

// B fragment: per-lane contiguous 16 halves, prepacked by prep kernel.
__device__ __forceinline__ v16h loadBfrag(const _Float16* wc, int halfOff, int lane) {
    FragU f;
    const uint4* p = (const uint4*)(wc + halfOff + lane * 16);
    f.q[0] = p[0];
    f.q[1] = p[1];
    return f.v;
}

// A fragment from LDS row-major activations (contiguous k order, same as the prepacked B):
// lane m = lane%16; halves j=0..15 -> K = kbase + (hi?16:0) + j.
__device__ __forceinline__ v16h loadAfrag(const _Float16* lds, int stride, int mA, int kbase, int hi) {
    FragU f;
    const int k1 = kbase + (hi ? 16 : 0);
    const int k2 = k1 + 8;
    f.q[0] = *(const uint4*)(lds + mA * stride + k1);
    f.q[1] = *(const uint4*)(lds + mA * stride + k2);
    return f.v;
}

// ---------------------------------------------------------------------------
// Prep: convert f32 weights to f16 WMMA B-fragments in d_ws.
// B-matrix (KxN=32x16) packing, contiguous k order matching loadAfrag: lane%16 = N column,
// halves j=0..15 hold K = (lane<16 ? 0 : 16) + j.
// ---------------------------------------------------------------------------
__global__ void rrsep_prep(const float* __restrict__ W1,
                           const float* __restrict__ W2,
                           const float* __restrict__ W3,
                           _Float16* __restrict__ wfrag)
{
    const int idx = blockIdx.x * 256 + threadIdx.x;
    const int total = 24 * COMBO_HALVES;
    if (idx >= total) return;
    const int combo = idx / COMBO_HALVES;
    const int r     = idx % COMBO_HALVES;
    const int s = combo >> 3;
    const int p = (combo >> 1) & 3;
    const int c = combo & 1;

    float val = 0.f;
    if (r < L2_OFF) {                       // layer1: W1 (24x64), K padded to 32
        const int t = r / 512, q = r % 512;
        const int lane = q / 16, j = q % 16;
        const int n = t * 16 + (lane & 15);
        const int k = ((lane < 16) ? 0 : 16) + j;
        if (k < 24) val = W1[ (((s*4 + p)*2 + c)*24 + k)*64 + n ];
    } else if (r < L3_OFF) {                // layer2: W2 (64x64), two K steps
        const int rr = r - L2_OFF;
        const int kstep = rr / 2048;
        const int t = (rr / 512) & 3;
        const int q = rr % 512;
        const int lane = q / 16, j = q % 16;
        const int n = t * 16 + (lane & 15);
        const int k = kstep * 32 + ((lane < 16) ? 0 : 16) + j;
        val = W2[ (((s*4 + p)*2 + c)*64 + k)*64 + n ];
    } else {                                // layer3: W3 (64x12), N padded to 16
        const int rr = r - L3_OFF;
        const int kstep = rr / 512;
        const int q = rr % 512;
        const int lane = q / 16, j = q % 16;
        const int n = lane & 15;
        const int k = kstep * 32 + ((lane < 16) ? 0 : 16) + j;
        if (n < 12) val = W3[ (((s*4 + p)*2 + c)*64 + k)*12 + n ];
    }
    wfrag[idx] = (_Float16)val;
}

// ---------------------------------------------------------------------------
// Main: one wave = 16 samples through all 3 stages / 4 ports / re+im MLPs.
// Activations in C/D layout regs; restaged via wave-private LDS for
// A-fragments. No workgroup barriers anywhere (waves fully independent).
// ---------------------------------------------------------------------------
__global__ __launch_bounds__(256) void rrsep_main(
    const float* __restrict__ yR, const float* __restrict__ yI,
    const float* __restrict__ b1, const float* __restrict__ b2,
    const float* __restrict__ b3,
    const _Float16* __restrict__ wfrag,
    float* __restrict__ out)
{
    __shared__ _Float16 ldsX[WAVES][16 * 40];   // x staging (cols 0..23 data, 24..31 zero)
    __shared__ _Float16 ldsH[WAVES][16 * 72];   // hidden staging (16x64, padded stride)

    const int tid  = threadIdx.x;
    const int wave = tid >> 5;
    const int lane = tid & 31;
    const int lcol = lane & 15;       // C/D layout: N column; A layout: M row
    const int hi   = lane >> 4;       // C/D layout: rows 8..15 half
    const long long base = (long long)blockIdx.x * BLK_SAMPLES + wave * TILE;

    _Float16* lx = ldsX[wave];
    _Float16* lh = ldsH[wave];

    // zero K-pad columns 24..31 of x buffer (written once, never touched again)
    for (int idx = lane; idx < 128; idx += 32) {
        const int m = idx >> 3;
        lx[m * 40 + 24 + (idx & 7)] = (_Float16)0.f;
    }
    lds_fence();

    // ---- load y into C/D layout, compute energy + normalize -------------
    float yrN[8], yiN[8], eArr[8];
    #pragma unroll
    for (int r = 0; r < 8; ++r) {
        const int m = hi * 8 + r;
        const long long row = base + m;
        float a = 0.f, b = 0.f;
        if (lcol < L_DIM) {
            a = yR[row * L_DIM + lcol];
            b = yI[row * L_DIM + lcol];
        }
        yrN[r] = a; yiN[r] = b;
    }
    #pragma unroll
    for (int r = 0; r < 8; ++r) {
        float ss = yrN[r] * yrN[r] + yiN[r] * yiN[r];
        ss += __shfl_xor(ss, 1, 32);
        ss += __shfl_xor(ss, 2, 32);
        ss += __shfl_xor(ss, 4, 32);
        ss += __shfl_xor(ss, 8, 32);
        const float e = sqrtf(ss * (1.0f / 12.0f));
        eArr[r] = e;
        const float inv = 1.0f / (e + EPSV);
        yrN[r] *= inv;
        yiN[r] *= inv;
    }

    float fr[P_DIM][8], fi[P_DIM][8];
    #pragma unroll
    for (int p = 0; p < P_DIM; ++p)
        #pragma unroll
        for (int r = 0; r < 8; ++r) { fr[p][r] = yrN[r]; fi[p][r] = yiN[r]; }

    // ---- stages ---------------------------------------------------------
    for (int s = 0; s < S_DIM; ++s) {
        float nf[2][P_DIM][8];   // [c][p][r] layer-3 outputs (pre-residual)

        #pragma unroll
        for (int p = 0; p < P_DIM; ++p) {
            // stage x = concat(fr_p, fi_p) into LDS (f16, row-major)
            lds_fence();
            #pragma unroll
            for (int r = 0; r < 8; ++r) {
                const int m = hi * 8 + r;
                if (lcol < L_DIM) {
                    lx[m * 40 + lcol]          = (_Float16)fr[p][r];
                    lx[m * 40 + L_DIM + lcol]  = (_Float16)fi[p][r];
                }
            }
            lds_fence();
            const v16h ax = loadAfrag(lx, 40, lcol, 0, hi);

            #pragma unroll
            for (int c = 0; c < 2; ++c) {
                const int combo = (s * P_DIM + p) * 2 + c;
                const _Float16* wc = wfrag + combo * COMBO_HALVES;
                const int bOff = combo * H_DIM;

                // ---- layer 1: (16x32) x (32x64) -------------------------
                v8f acc[4];
                #pragma unroll
                for (int t = 0; t < 4; ++t) {
                    const v16h bf = loadBfrag(wc, L1_OFF + t * 512, lane);
                    v8f z = {};
                    acc[t] = __builtin_amdgcn_wmma_f32_16x16x32_f16(
                        false, ax, false, bf, (short)0, z, false, false);
                }
                lds_fence();
                #pragma unroll
                for (int t = 0; t < 4; ++t) {
                    const float bv = b1[bOff + t * 16 + lcol];
                    #pragma unroll
                    for (int r = 0; r < 8; ++r) {
                        const int m = hi * 8 + r;
                        lh[m * 72 + t * 16 + lcol] =
                            (_Float16)fmaxf(acc[t][r] + bv, 0.f);
                    }
                }
                lds_fence();
                v16h a0 = loadAfrag(lh, 72, lcol, 0, hi);
                v16h a1 = loadAfrag(lh, 72, lcol, 32, hi);

                // ---- layer 2: (16x64) x (64x64) -------------------------
                #pragma unroll
                for (int t = 0; t < 4; ++t) {
                    const v16h b0 = loadBfrag(wc, L2_OFF + (0 * 4 + t) * 512, lane);
                    const v16h b1f = loadBfrag(wc, L2_OFF + (1 * 4 + t) * 512, lane);
                    v8f z = {};
                    v8f d = __builtin_amdgcn_wmma_f32_16x16x32_f16(
                        false, a0, false, b0, (short)0, z, false, false);
                    acc[t] = __builtin_amdgcn_wmma_f32_16x16x32_f16(
                        false, a1, false, b1f, (short)0, d, false, false);
                }
                lds_fence();
                #pragma unroll
                for (int t = 0; t < 4; ++t) {
                    const float bv = b2[bOff + t * 16 + lcol];
                    #pragma unroll
                    for (int r = 0; r < 8; ++r) {
                        const int m = hi * 8 + r;
                        lh[m * 72 + t * 16 + lcol] =
                            (_Float16)fmaxf(acc[t][r] + bv, 0.f);
                    }
                }
                lds_fence();
                const v16h c0 = loadAfrag(lh, 72, lcol, 0, hi);
                const v16h c1 = loadAfrag(lh, 72, lcol, 32, hi);

                // ---- layer 3: (16x64) x (64x16[12]) ---------------------
                const v16h w0 = loadBfrag(wc, L3_OFF + 0 * 512, lane);
                const v16h w1 = loadBfrag(wc, L3_OFF + 1 * 512, lane);
                v8f z = {};
                v8f d = __builtin_amdgcn_wmma_f32_16x16x32_f16(
                    false, c0, false, w0, (short)0, z, false, false);
                d = __builtin_amdgcn_wmma_f32_16x16x32_f16(
                    false, c1, false, w1, (short)0, d, false, false);
                const float b3v = (lcol < L_DIM) ? b3[combo * L_DIM + lcol] : 0.f;
                #pragma unroll
                for (int r = 0; r < 8; ++r) nf[c][p][r] = d[r] + b3v;
            } // c
        } // p

        // ---- residual coupling across ports (elementwise in C/D layout) -
        #pragma unroll
        for (int r = 0; r < 8; ++r) {
            const float sr = nf[0][0][r] + nf[0][1][r] + nf[0][2][r] + nf[0][3][r];
            const float si = nf[1][0][r] + nf[1][1][r] + nf[1][2][r] + nf[1][3][r];
            const float rr = yrN[r] - sr;
            const float ri = yiN[r] - si;
            #pragma unroll
            for (int p = 0; p < P_DIM; ++p) {
                fr[p][r] = nf[0][p][r] + rr;
                fi[p][r] = nf[1][p][r] + ri;
            }
        }
    } // stages

    // ---- rescale by energy, write (B, P, L, 2) --------------------------
    #pragma unroll
    for (int p = 0; p < P_DIM; ++p) {
        #pragma unroll
        for (int r = 0; r < 8; ++r) {
            if (lcol < L_DIM) {
                const int m = hi * 8 + r;
                const long long bidx = base + m;
                float2 o;
                o.x = fr[p][r] * eArr[r];
                o.y = fi[p][r] * eArr[r];
                *(float2*)(out + (((bidx * P_DIM) + p) * L_DIM + lcol) * 2) = o;
            }
        }
    }
}

extern "C" void kernel_launch(void* const* d_in, const int* in_sizes, int n_in,
                              void* d_out, int out_size, void* d_ws, size_t ws_size,
                              hipStream_t stream) {
    const float* yR = (const float*)d_in[0];
    const float* yI = (const float*)d_in[1];
    const float* W1 = (const float*)d_in[2];
    const float* b1 = (const float*)d_in[3];
    const float* W2 = (const float*)d_in[4];
    const float* b2 = (const float*)d_in[5];
    const float* W3 = (const float*)d_in[6];
    const float* b3 = (const float*)d_in[7];
    _Float16* wfrag = (_Float16*)d_ws;   // needs 24*7168*2 = 344,064 bytes
    float* out = (float*)d_out;

    const int prepTotal = 24 * COMBO_HALVES;
    rrsep_prep<<<(prepTotal + 255) / 256, 256, 0, stream>>>(W1, W2, W3, wfrag);

    const int nBlocks = B_TOT / BLK_SAMPLES;   // 2048
    rrsep_main<<<nBlocks, 256, 0, stream>>>(yR, yI, b1, b2, b3, wfrag, out);
}
